// GatedGCN_backwards_77343771066509
// MI455X (gfx1250) — matrix operand order repaired and checked by reading, not compiled, not run on hardware
//
#include <hip/hip_runtime.h>
#include <hip/hip_bf16.h>

#define N_NODES 100000
#define N_EDGES 1250000
#define DIM 64

typedef __attribute__((ext_vector_type(16))) __bf16 v16bf;
typedef __attribute__((ext_vector_type(8)))  float  v8f;

__device__ __forceinline__ __bf16 f2bf(float x) { return (__bf16)x; }

// A-matrix 16x32 bf16 fragment (ISA 7.12.2): lane L holds row M = L%16;
// VGPR packing: half = L/16 -> K = half*8 + j (j<8), 16 + half*8 + (j-8) (j>=8)
template <bool NT>
__device__ __forceinline__ v16bf load_a_frag(const float* __restrict__ row, int lane, int kk) {
  const int half = lane >> 4;
  v16bf a;
#pragma unroll
  for (int j = 0; j < 16; ++j) {
    int k = kk * 32 + ((j < 8) ? (half * 8 + j) : (16 + half * 8 + (j - 8)));
    float x = NT ? __builtin_nontemporal_load(&row[k]) : row[k];
    a[j] = f2bf(x);
  }
  return a;
}

// B-matrix 32x16 bf16 fragment: lane L holds column N = L%16,
// K = half*8 + j (j<8), 16 + half*8 + (j-8) (j>=8).  Wt is stored [k][n] (already transposed).
__device__ __forceinline__ v16bf load_b_frag(const __bf16* __restrict__ Wt, int lane, int kk, int nt) {
  const int half = lane >> 4;
  const __bf16* p = Wt + kk * 32 * DIM + nt * 16 + (lane & 15);
  v16bf b;
#pragma unroll
  for (int j = 0; j < 16; ++j) b[j] = p[((j < 8) ? (half * 8 + j) : (16 + half * 8 + (j - 8))) * DIM];
  return b;
}

__global__ void __launch_bounds__(256) zero_f32(float* __restrict__ p, long long n) {
  long long i = (long long)blockIdx.x * blockDim.x + threadIdx.x;
  const long long stride = (long long)gridDim.x * blockDim.x;
  for (; i < n; i += stride) p[i] = 0.0f;
}

// ---- fused 4-weight node GEMM: O_w = h @ W_w^T + b_w, WMMA bf16 ----
__global__ void __launch_bounds__(256)
node_gemm4(const float* __restrict__ h,
           const float* __restrict__ W0, const float* __restrict__ b0,
           const float* __restrict__ W1, const float* __restrict__ b1,
           const float* __restrict__ W2, const float* __restrict__ b2,
           const float* __restrict__ W3, const float* __restrict__ b3,
           float* __restrict__ O0, float* __restrict__ O1,
           float* __restrict__ O2, float* __restrict__ O3) {
  __shared__ __bf16 Wt[4][DIM * DIM];   // transposed: Wt[w][k*64+n] = W_w[n][k]
  __shared__ float  Bs[4][DIM];
  const float* Ws[4] = {W0, W1, W2, W3};
  const float* bs[4] = {b0, b1, b2, b3};
  float* Os[4] = {O0, O1, O2, O3};
#pragma unroll
  for (int w = 0; w < 4; ++w) {
    for (int i = threadIdx.x; i < DIM * DIM; i += 256) {
      int n = i >> 6, k = i & 63;
      Wt[w][k * DIM + n] = f2bf(Ws[w][i]);
    }
    if (threadIdx.x < DIM) Bs[w][threadIdx.x] = bs[w][threadIdx.x];
  }
  __syncthreads();

  const int wave = threadIdx.x >> 5, lane = threadIdx.x & 31;
  const int tile = blockIdx.x * 8 + wave;
  if (tile >= N_NODES / 16) return;
  const int m0 = tile * 16;
  const float* rowp = h + (size_t)(m0 + (lane & 15)) * DIM;
  const v16bf a0 = load_a_frag<false>(rowp, lane, 0);
  const v16bf a1 = load_a_frag<false>(rowp, lane, 1);
  const int n_lane = lane & 15, mbase = (lane >> 4) * 8;

#pragma unroll
  for (int w = 0; w < 4; ++w) {
#pragma unroll
    for (int nt = 0; nt < 4; ++nt) {
      const float bias = Bs[w][nt * 16 + n_lane];
      v8f c;
#pragma unroll
      for (int v = 0; v < 8; ++v) c[v] = bias;
      v16bf bf0 = load_b_frag(Wt[w], lane, 0, nt);
      v16bf bf1 = load_b_frag(Wt[w], lane, 1, nt);
      c = __builtin_amdgcn_wmma_f32_16x16x32_bf16(false, a0, false, bf0, (short)0, c, false, false);
      c = __builtin_amdgcn_wmma_f32_16x16x32_bf16(false, a1, false, bf1, (short)0, c, false, false);
      float* outp = Os[w] + (size_t)m0 * DIM + nt * 16 + n_lane;
#pragma unroll
      for (int v = 0; v < 8; ++v) outp[(size_t)(mbase + v) * DIM] = c[v];
    }
  }
}

// ---- edge GEMM fused with gather AND the edge-BN column statistics ----
// epre = e @ B3^T + b3 + B1h[src] + B2h[dst]   (NT-streamed to the output region)
// acc[0:64] += column sums, acc[64:128] += column sums of squares
__global__ void __launch_bounds__(256)
edge_gemm(const float* __restrict__ e, const float* __restrict__ W, const float* __restrict__ bb,
          const int* __restrict__ src, const int* __restrict__ dst,
          const float* __restrict__ B1h, const float* __restrict__ B2h,
          float* __restrict__ epre, float* __restrict__ acc) {
  __shared__ __bf16 Wt[DIM * DIM];
  __shared__ float  Bs[DIM];
  __shared__ float  sacc[128];   // block-level BN partial sums
  for (int i = threadIdx.x; i < DIM * DIM; i += 256) {
    int n = i >> 6, k = i & 63;
    Wt[k * DIM + n] = f2bf(W[i]);
  }
  if (threadIdx.x < DIM) Bs[threadIdx.x] = bb[threadIdx.x];
  if (threadIdx.x < 128) sacc[threadIdx.x] = 0.0f;
  __syncthreads();

  const int wave = threadIdx.x >> 5, lane = threadIdx.x & 31;
  const int tile = blockIdx.x * 8 + wave;
  const int n_lane = lane & 15, mbase = (lane >> 4) * 8;

  if (tile < N_EDGES / 16) {
    const int m0 = tile * 16;
    const float* rowp = e + (size_t)(m0 + (lane & 15)) * DIM;
    __builtin_prefetch(rowp + 128 * DIM, 0, 1);   // stream-ahead hint (global_prefetch_b8)
    const v16bf a0 = load_a_frag<true>(rowp, lane, 0);
    const v16bf a1 = load_a_frag<true>(rowp, lane, 1);

#pragma unroll
    for (int nt = 0; nt < 4; ++nt) {
      const float bias = Bs[nt * 16 + n_lane];
      v8f c;
#pragma unroll
      for (int v = 0; v < 8; ++v) c[v] = bias;
      v16bf bf0 = load_b_frag(Wt, lane, 0, nt);
      v16bf bf1 = load_b_frag(Wt, lane, 1, nt);
      c = __builtin_amdgcn_wmma_f32_16x16x32_bf16(false, a0, false, bf0, (short)0, c, false, false);
      c = __builtin_amdgcn_wmma_f32_16x16x32_bf16(false, a1, false, bf1, (short)0, c, false, false);
      const int col = nt * 16 + n_lane;
      float s = 0.0f, s2 = 0.0f;
#pragma unroll
      for (int v = 0; v < 8; ++v) {
        const int em = m0 + mbase + v;
        const int sN = src[em], dN = dst[em];
        const float val = c[v] + B1h[(size_t)sN * DIM + col] + B2h[(size_t)dN * DIM + col];
        __builtin_nontemporal_store(val, &epre[(size_t)em * DIM + col]);
        s += val; s2 += val * val;
      }
      atomicAdd(&sacc[col], s);            // ds_add_f32
      atomicAdd(&sacc[64 + col], s2);
    }
  }
  __syncthreads();
  if (threadIdx.x < 128) atomicAdd(&acc[threadIdx.x], sacc[threadIdx.x]);
}

// ---- edge BN+ReLU+residual (in place) + sigmoid gating + segment atomics ----
__global__ void __launch_bounds__(256)
edge_finalize(float* __restrict__ e_io, const float* __restrict__ e_in,
              const int* __restrict__ src, const int* __restrict__ dst,
              const float* __restrict__ A3h,
              const float* __restrict__ g, const float* __restrict__ b,
              const float* __restrict__ acc,
              float* __restrict__ ssh, float* __restrict__ ss) {
  const long long idx = (long long)blockIdx.x * blockDim.x + threadIdx.x;
  if (idx >= (long long)N_EDGES * DIM) return;
  const int col = (int)(idx & 63);
  const long long ei = idx >> 6;
  const float inv = 1.0f / (float)N_EDGES;
  const float mean = acc[col] * inv;
  const float var = acc[64 + col] * inv - mean * mean;
  const float pre = __builtin_nontemporal_load(&e_io[idx]);
  float y = g[col] * (pre - mean) * rsqrtf(var + 1e-5f) + b[col];
  y = fmaxf(y, 0.0f) + __builtin_nontemporal_load(&e_in[idx]);
  __builtin_nontemporal_store(y, &e_io[idx]);   // final e_ik output
  const float sig = 1.0f / (1.0f + __expf(-y));
  const int s = src[ei], d = dst[ei];
  atomicAdd(&ss[(size_t)s * DIM + col], sig);
  atomicAdd(&ssh[(size_t)s * DIM + col], A3h[(size_t)d * DIM + col] * sig);
}

// ---- node h_tmp = A1h + ssh/(ss+1e-6): column stats ----
__global__ void __launch_bounds__(256)
node_stats(const float* __restrict__ A1h, const float* __restrict__ ssh,
           const float* __restrict__ ss, float* __restrict__ acc) {
  const int col = threadIdx.x & 63;
  const long long rgrp = (threadIdx.x >> 6) + ((long long)blockIdx.x << 2);
  const long long rstride = (long long)gridDim.x << 2;
  float s = 0.f, s2 = 0.f;
  for (long long r = rgrp; r < N_NODES; r += rstride) {
    const long long i = r * DIM + col;
    float x = A1h[i] + ssh[i] / (ss[i] + 1e-6f);
    s += x; s2 += x * x;
  }
  __shared__ float ls[256], ls2[256];
  ls[threadIdx.x] = s; ls2[threadIdx.x] = s2;
  __syncthreads();
  if (threadIdx.x < 64) {
    float t  = ls[threadIdx.x]  + ls[threadIdx.x + 64]  + ls[threadIdx.x + 128]  + ls[threadIdx.x + 192];
    float t2 = ls2[threadIdx.x] + ls2[threadIdx.x + 64] + ls2[threadIdx.x + 128] + ls2[threadIdx.x + 192];
    atomicAdd(&acc[col], t);
    atomicAdd(&acc[64 + col], t2);
  }
}

__global__ void __launch_bounds__(256)
node_finalize(const float* __restrict__ A1h, const float* __restrict__ ssh,
              const float* __restrict__ ss, const float* __restrict__ h,
              const float* __restrict__ g, const float* __restrict__ b,
              const float* __restrict__ acc, float* __restrict__ h_out) {
  const long long idx = (long long)blockIdx.x * blockDim.x + threadIdx.x;
  if (idx >= (long long)N_NODES * DIM) return;
  const int col = (int)(idx & 63);
  const float inv = 1.0f / (float)N_NODES;
  const float mean = acc[col] * inv;
  const float var = acc[64 + col] * inv - mean * mean;
  const float x = A1h[idx] + ssh[idx] / (ss[idx] + 1e-6f);
  float y = g[col] * (x - mean) * rsqrtf(var + 1e-5f) + b[col];
  __builtin_nontemporal_store(fmaxf(y, 0.0f) + h[idx], &h_out[idx]);
}

extern "C" void kernel_launch(void* const* d_in, const int* in_sizes, int n_in,
                              void* d_out, int out_size, void* d_ws, size_t ws_size,
                              hipStream_t stream) {
  const float* h    = (const float*)d_in[0];
  const float* e    = (const float*)d_in[1];
  const int*   src  = (const int*)d_in[2];
  const int*   dst  = (const int*)d_in[3];
  const float* A1_w = (const float*)d_in[4];
  const float* A1_b = (const float*)d_in[5];
  const float* A3_w = (const float*)d_in[6];
  const float* A3_b = (const float*)d_in[7];
  const float* B1_w = (const float*)d_in[8];
  const float* B1_b = (const float*)d_in[9];
  const float* B2_w = (const float*)d_in[10];
  const float* B2_b = (const float*)d_in[11];
  const float* B3_w = (const float*)d_in[12];
  const float* B3_b = (const float*)d_in[13];
  const float* bn_h_g = (const float*)d_in[14];
  const float* bn_h_b = (const float*)d_in[15];
  const float* bn_e_g = (const float*)d_in[16];
  const float* bn_e_b = (const float*)d_in[17];

  const size_t nd = (size_t)N_NODES * DIM;
  float* A1h = (float*)d_ws;
  float* A3h = A1h + nd;
  float* B1h = A3h + nd;
  float* B2h = B1h + nd;
  float* ssh = B2h + nd;       // segment sum of A3h[dst]*sigma
  float* ss  = ssh + nd;       // segment sum of sigma
  float* acc_e = ss + nd;      // 128 floats (edge BN sum/sumsq)
  float* acc_h = acc_e + 128;  // 128 floats (node BN sum/sumsq)

  float* h_out = (float*)d_out;               // [N_NODES, 64]
  float* e_out = h_out + nd;                  // [N_EDGES, 64] (pre-BN, then final)

  // 1. zero the atomic accumulators (ssh, ss, acc_e, acc_h are contiguous)
  zero_f32<<<2048, 256, 0, stream>>>(ssh, (long long)(2 * nd + 256));

  // 2. four node GEMMs via WMMA
  node_gemm4<<<(N_NODES / 16 + 7) / 8, 256, 0, stream>>>(
      h, A1_w, A1_b, A3_w, A3_b, B1_w, B1_b, B2_w, B2_b, A1h, A3h, B1h, B2h);

  // 3. edge GEMM + gather + fused BN statistics (pre-BN e -> output region, NT)
  edge_gemm<<<(N_EDGES / 16 + 7) / 8, 256, 0, stream>>>(
      e, B3_w, B3_b, src, dst, B1h, B2h, e_out, acc_e);

  // 4. edge finalize + sigmoid gating + segment sums (atomics land in L2)
  edge_finalize<<<(int)(((long long)N_EDGES * DIM) / 256), 256, 0, stream>>>(
      e_out, e, src, dst, A3h, bn_e_g, bn_e_b, acc_e, ssh, ss);

  // 5. node BN statistics over h_tmp (computed on the fly)
  node_stats<<<512, 256, 0, stream>>>(A1h, ssh, ss, acc_h);

  // 6. node finalize
  node_finalize<<<(int)(nd / 256), 256, 0, stream>>>(
      A1h, ssh, ss, h, bn_h_g, bn_h_b, acc_h, h_out);
}
